// ClassifierV2_46231027974393
// MI455X (gfx1250) — matrix operand order repaired and checked by reading, not compiled, not run on hardware
//
#include <hip/hip_runtime.h>
#include <hip/hip_bf16.h>
#include <cstdint>
#include <cstddef>

// ---------------- problem constants (from reference) ----------------
#define DD      256        // in features / hidden (H*C)
#define HH      4          // heads
#define CC      64         // channels per head (hidden layers)
#define OUTC    5          // classes
#define KDIM    256        // GEMM K
#define CAT_LD  768        // concatenated [Wl|Wr|Wlin] output width (hidden layers)
#define FIN_LD  48         // concatenated final width: 20 + 20 + 5 -> padded 48
#define AG_LD   20         // final aggregation width H*OUT
#define NEG_SLOPE 0.2f

typedef __bf16 bf16_t;
typedef __attribute__((ext_vector_type(16))) __bf16 v16bf;
typedef __attribute__((ext_vector_type(8)))  __bf16 v8bf;
typedef __attribute__((ext_vector_type(8)))  float  v8f;
typedef __attribute__((ext_vector_type(4)))  float  v4f;

// ================= utility kernels =================
__global__ void fill_f32(float* __restrict__ p, long n, float v) {
  long i = (long)blockIdx.x * blockDim.x + threadIdx.x;
  if (i < n) p[i] = v;
}

__global__ void copy_f32(const float* __restrict__ s, float* __restrict__ d, long n) {
  long i = (long)blockIdx.x * blockDim.x + threadIdx.x;
  if (i < n) d[i] = s[i];
}

__global__ void f32_to_bf16(const float* __restrict__ x, bf16_t* __restrict__ y, long n) {
  long i = (long)blockIdx.x * blockDim.x + threadIdx.x;
  if (i < n) y[i] = (bf16_t)x[i];
}

// Transpose fp32 weight W[K, Nout] (row-major) into bf16 WT[NoutPad, K],
// zero-padding rows Nout..NoutPad-1 so GEMM tiles are always full.
__global__ void prep_wt(const float* __restrict__ W, bf16_t* __restrict__ WT,
                        int K, int Nout, int NoutPad) {
  long i = (long)blockIdx.x * blockDim.x + threadIdx.x;
  long n = (long)NoutPad * K;
  if (i >= n) return;
  int r = (int)(i / K);
  int k = (int)(i % K);
  WT[i] = (r < Nout) ? (bf16_t)W[(size_t)k * Nout + r] : (bf16_t)0.0f;
}

// ================= bf16 WMMA GEMM =================
// Y[M, Np] = A[M, K](bf16) @ WT[Np, K]^T (bf16)  + bias[Np]   (fp32 out)
// One wave computes a 16 x (16*NB) tile: A fragment loaded once per k-step,
// reused across NB WMMAs (4x less A traffic for NB=4).
template <int NB>
__global__ __launch_bounds__(256)
void gemm_bf16_wmma(const bf16_t* __restrict__ A,
                    const bf16_t* __restrict__ WT,
                    const float*  __restrict__ bias,
                    float* __restrict__ Y,
                    int M, int K, int Np) {
  const int lane    = threadIdx.x & 31;
  const int wave    = threadIdx.x >> 5;
  const int groupsN = (Np >> 4) / NB;
  const int g       = blockIdx.x * 8 + wave;
  if (g >= (M >> 4) * groupsN) return;   // uniform per wave: EXEC all-1 inside
  const int tm   = g / groupsN;
  const int tg   = g % groupsN;
  const int half = lane >> 4;
  const int l15  = lane & 15;

  const bf16_t* arow = A + (size_t)(tm * 16 + l15) * K;
  const bf16_t* brow[NB];
  #pragma unroll
  for (int j = 0; j < NB; ++j)
    brow[j] = WT + (size_t)((tg * NB + j) * 16 + l15) * K;

  v8f acc[NB];
  #pragma unroll
  for (int j = 0; j < NB; ++j) acc[j] = {};

  for (int k = 0; k < K; k += 32) {
    // CDNA5 16-bit A 16x32 layout: lanes 0-15 hold K = k+0..7, k+16..23;
    // lanes 16-31 hold K = k+8..15, k+24..31 (same rows M=l15).
    v8bf alo = *reinterpret_cast<const v8bf*>(arow + k + half * 8);
    v8bf ahi = *reinterpret_cast<const v8bf*>(arow + k + 16 + half * 8);
    v16bf a;
    #pragma unroll
    for (int t = 0; t < 8; ++t) { a[t] = alo[t]; a[8 + t] = ahi[t]; }
    #pragma unroll
    for (int j = 0; j < NB; ++j) {
      // B 32x16: lane holds one column (= row of WT); same K order as A:
      // lanes 0-15 K = k+0..7, k+16..23; lanes 16-31 K = k+8..15, k+24..31.
      v8bf blo = *reinterpret_cast<const v8bf*>(brow[j] + k + half * 8);
      v8bf bhi = *reinterpret_cast<const v8bf*>(brow[j] + k + 16 + half * 8);
      v16bf b;
      #pragma unroll
      for (int t = 0; t < 8; ++t) { b[t] = blo[t]; b[8 + t] = bhi[t]; }
      acc[j] = __builtin_amdgcn_wmma_f32_16x16x32_bf16(false, a, false, b,
                                                       (short)0, acc[j], false, false);
    }
  }
  // D layout: lane&15 = column; VGPR v -> row v + 8*(lane>=16)
  #pragma unroll
  for (int j = 0; j < NB; ++j) {
    const int col = (tg * NB + j) * 16 + l15;
    const float bv = bias[col];
    #pragma unroll
    for (int v = 0; v < 8; ++v) {
      int row = tm * 16 + half * 8 + v;
      Y[(size_t)row * Np + col] = acc[j][v] + bv;
    }
  }
}

// ================= edge-phase kernels =================
__device__ __forceinline__ void atomicMaxF(float* addr, float val) {
  // monotone bit trick: signed max for >=0, unsigned min for <0
  if (val >= 0.0f) atomicMax((int*)addr, __float_as_int(val));
  else             atomicMin((unsigned int*)addr, __float_as_uint(val));
}

// ---- hidden layers (H=4, C=64): one wave per edge, fully coalesced ----
// lane owns 8 contiguous channels (32B); head = lane>>3; per-head dot product
// reduced across the 8 lanes of each head group via shfl_xor.
__global__ __launch_bounds__(256)
void edge_score_w(const int* __restrict__ src, const int* __restrict__ dst,
                  const float* __restrict__ Y, int ldY, int xrOff,
                  const float* __restrict__ att,
                  float* __restrict__ score, long E) {
  const int lane = threadIdx.x & 31;
  const long e = (long)blockIdx.x * 8 + (threadIdx.x >> 5);
  if (e >= E) return;
  const int s = src[e], d = dst[e];
  const int c0 = lane * 8;                 // channel base within H*C=256
  const float* pl = Y + (size_t)s * ldY + c0;
  const float* pr = Y + (size_t)d * ldY + xrOff + c0;
  const float* pa = att + c0;              // att laid out [H, C] flat = 256
  v4f l0 = *reinterpret_cast<const v4f*>(pl);
  v4f l1 = *reinterpret_cast<const v4f*>(pl + 4);
  v4f r0 = *reinterpret_cast<const v4f*>(pr);
  v4f r1 = *reinterpret_cast<const v4f*>(pr + 4);
  float part = 0.0f;
  #pragma unroll
  for (int t = 0; t < 4; ++t) {
    float v = l0[t] + r0[t]; v = v > 0.0f ? v : NEG_SLOPE * v; part += pa[t] * v;
    float w = l1[t] + r1[t]; w = w > 0.0f ? w : NEG_SLOPE * w; part += pa[4 + t] * w;
  }
  part += __shfl_xor(part, 1);
  part += __shfl_xor(part, 2);
  part += __shfl_xor(part, 4);             // now all 8 lanes of a head hold the sum
  if ((lane & 7) == 0) score[e * HH + (lane >> 3)] = part;
}

__global__ __launch_bounds__(256)
void edge_aggregate_w(const int* __restrict__ src, const int* __restrict__ dst,
                      const float* __restrict__ Y, int ldY,
                      const float* __restrict__ escore,
                      const float* __restrict__ denom,
                      float* __restrict__ outp, long E) {
  const int lane = threadIdx.x & 31;
  const long e = (long)blockIdx.x * 8 + (threadIdx.x >> 5);
  if (e >= E) return;
  const int s = src[e], d = dst[e];
  const int h = lane >> 3;
  const int c0 = lane * 8;
  const float alpha = escore[e * HH + h] /
                      (denom[(size_t)d * HH + h] + 1e-16f);
  const float* pl = Y + (size_t)s * ldY + c0;
  float* po = outp + (size_t)d * DD + c0;
  v4f l0 = *reinterpret_cast<const v4f*>(pl);
  v4f l1 = *reinterpret_cast<const v4f*>(pl + 4);
  #pragma unroll
  for (int t = 0; t < 4; ++t) {
    atomicAdd(&po[t],     alpha * l0[t]);
    atomicAdd(&po[4 + t], alpha * l1[t]);
  }
}

// ---- generic per-(e,h) kernels (used for the small final layer, C=5) ----
__global__ void edge_score(const int* __restrict__ src, const int* __restrict__ dst,
                           const float* __restrict__ Y, int ldY, int xlOff, int xrOff,
                           const float* __restrict__ att,
                           float* __restrict__ score, long E, int H, int C) {
  long idx = (long)blockIdx.x * blockDim.x + threadIdx.x;
  if (idx >= E * H) return;
  long e = idx / H;
  int  h = (int)(idx % H);
  int  s = src[e], d = dst[e];
  const float* pl = Y + (size_t)s * ldY + xlOff + h * C;
  const float* pr = Y + (size_t)d * ldY + xrOff + h * C;
  const float* pa = att + h * C;
  float acc = 0.0f;
  for (int c = 0; c < C; ++c) {
    float v = pl[c] + pr[c];
    v = v > 0.0f ? v : NEG_SLOPE * v;
    acc += pa[c] * v;
  }
  score[idx] = acc;
}

__global__ void edge_segmax(const int* __restrict__ dst,
                            const float* __restrict__ score,
                            float* __restrict__ mmax, long E, int H) {
  long idx = (long)blockIdx.x * blockDim.x + threadIdx.x;
  if (idx >= E * H) return;
  long e = idx / H;
  int  h = (int)(idx % H);
  atomicMaxF(&mmax[(size_t)dst[e] * H + h], score[idx]);
}

__global__ void edge_expsum(const int* __restrict__ dst,
                            float* __restrict__ score,        // in: score, out: exp(score-m)
                            const float* __restrict__ mmax,
                            float* __restrict__ denom, long E, int H) {
  long idx = (long)blockIdx.x * blockDim.x + threadIdx.x;
  if (idx >= E * H) return;
  long e = idx / H;
  int  h = (int)(idx % H);
  int  d = dst[e];
  float a = __expf(score[idx] - mmax[(size_t)d * H + h]);
  score[idx] = a;
  atomicAdd(&denom[(size_t)d * H + h], a);
}

__global__ void edge_aggregate(const int* __restrict__ src, const int* __restrict__ dst,
                               const float* __restrict__ Y, int ldY, int xlOff,
                               const float* __restrict__ escore,
                               const float* __restrict__ denom,
                               float* __restrict__ outp, int outLd,
                               long E, int H, int C) {
  long idx = (long)blockIdx.x * blockDim.x + threadIdx.x;
  if (idx >= E * H) return;
  long e = idx / H;
  int  h = (int)(idx % H);
  int  s = src[e], d = dst[e];
  float alpha = escore[idx] / (denom[(size_t)d * H + h] + 1e-16f);
  const float* pl = Y + (size_t)s * ldY + xlOff + h * C;
  float* po = outp + (size_t)d * outLd + h * C;
  for (int c = 0; c < C; ++c) atomicAdd(&po[c], alpha * pl[c]);
}

// ================= GraphNorm =================
#define GN_ROWS 64
// thread t = channel t (coalesced across a block); each block covers 64 rows
__global__ __launch_bounds__(256)
void gn_stats(const float* __restrict__ cp, const float* __restrict__ bconv,
              float* __restrict__ sum, float* __restrict__ sumsq, int N) {
  int c  = threadIdx.x;          // 0..255
  int r0 = blockIdx.x * GN_ROWS;
  int r1 = min(r0 + GN_ROWS, N);
  float bc = bconv[c];
  float s1 = 0.0f, s2 = 0.0f;
  for (int r = r0; r < r1; ++r) {
    float v = cp[(size_t)r * DD + c] + bc;
    s1 += v; s2 += v * v;
  }
  atomicAdd(&sum[c],   s1);
  atomicAdd(&sumsq[c], s2);
}

// norm + add linear-skip + ELU, and emit next layer's bf16 activations
__global__ void gn_finalize(const float* __restrict__ cp, const float* __restrict__ bconv,
                            const float* __restrict__ sum, const float* __restrict__ sumsq,
                            const float* __restrict__ gw, const float* __restrict__ gb,
                            const float* __restrict__ gms,
                            const float* __restrict__ lp, int lpLd,
                            bf16_t* __restrict__ xb, int N) {
  long idx = (long)blockIdx.x * blockDim.x + threadIdx.x;
  if (idx >= (long)N * DD) return;
  int r = (int)(idx / DD);
  int c = (int)(idx % DD);
  float invN = 1.0f / (float)N;
  float mean = sum[c] * invN;
  float mu   = mean * gms[c];
  float var  = sumsq[c] * invN - 2.0f * mu * mean + mu * mu;
  float v    = cp[idx] + bconv[c] - mu;
  float y    = gw[c] * v * rsqrtf(var + 1e-5f) + gb[c];
  float t    = y + lp[(size_t)r * lpLd + c];
  float e    = t > 0.0f ? t : (__expf(t) - 1.0f);
  xb[idx] = (bf16_t)e;
}

// ================= final head: head-mean + linear + log_softmax =================
__global__ void final_head(const float* __restrict__ aggf,   // [N, H*OUT]
                           const float* __restrict__ fbconv, // [OUT]
                           const float* __restrict__ Yf,     // lp at col 40, ld 48
                           float* __restrict__ out, int N) {
  int n = blockIdx.x * blockDim.x + threadIdx.x;
  if (n >= N) return;
  float logit[OUTC];
  float mx = -3.0e38f;
  #pragma unroll
  for (int c = 0; c < OUTC; ++c) {
    float m = 0.0f;
    #pragma unroll
    for (int h = 0; h < HH; ++h) m += aggf[(size_t)n * AG_LD + h * OUTC + c];
    m = m * 0.25f + fbconv[c] + Yf[(size_t)n * FIN_LD + 40 + c];
    logit[c] = m;
    mx = fmaxf(mx, m);
  }
  float s = 0.0f;
  #pragma unroll
  for (int c = 0; c < OUTC; ++c) s += __expf(logit[c] - mx);
  float lse = __logf(s);
  #pragma unroll
  for (int c = 0; c < OUTC; ++c) out[(size_t)n * OUTC + c] = logit[c] - mx - lse;
}

// ================= host launcher =================
static inline long cdivl(long a, long b) { return (a + b - 1) / b; }

extern "C" void kernel_launch(void* const* d_in, const int* in_sizes, int n_in,
                              void* d_out, int out_size, void* d_ws, size_t ws_size,
                              hipStream_t stream) {
  const float* x      = (const float*)d_in[0];
  const int*   edge   = (const int*)d_in[1];
  const float* Wl     = (const float*)d_in[2];
  const float* Wr     = (const float*)d_in[3];
  const float* att    = (const float*)d_in[4];
  const float* bconv  = (const float*)d_in[5];
  const float* Wlin   = (const float*)d_in[6];
  const float* blin   = (const float*)d_in[7];
  const float* gn_w   = (const float*)d_in[8];
  const float* gn_b   = (const float*)d_in[9];
  const float* gn_ms  = (const float*)d_in[10];
  const float* fWl    = (const float*)d_in[11];
  const float* fWr    = (const float*)d_in[12];
  const float* fatt   = (const float*)d_in[13];
  const float* fbconv = (const float*)d_in[14];
  const float* fWlin  = (const float*)d_in[15];
  const float* fblin  = (const float*)d_in[16];
  float* out = (float*)d_out;

  const long N = in_sizes[0] / DD;         // 50000
  const long E = in_sizes[1] / 2;          // 800000
  const int* src = edge;
  const int* dst = edge + E;

  // ---- workspace bump allocator ----
  size_t off = 0;
  auto alloc = [&](size_t bytes) -> char* {
    char* p = (char*)d_ws + off;
    off += (bytes + 255) & ~(size_t)255;
    return p;
  };
  bf16_t* xb      = (bf16_t*)alloc((size_t)N * DD * sizeof(bf16_t));
  bf16_t* wtcat0  = (bf16_t*)alloc((size_t)CAT_LD * KDIM * sizeof(bf16_t));
  bf16_t* wtcat1  = (bf16_t*)alloc((size_t)CAT_LD * KDIM * sizeof(bf16_t));
  bf16_t* wtf     = (bf16_t*)alloc((size_t)FIN_LD * KDIM * sizeof(bf16_t));
  float*  bias    = (float*)alloc((size_t)(CAT_LD * 2 + FIN_LD) * sizeof(float));
  float*  bcat0   = bias;
  float*  bcat1   = bias + CAT_LD;
  float*  bcatf   = bias + 2 * CAT_LD;
  float*  Ycat    = (float*)alloc((size_t)N * CAT_LD * sizeof(float));
  float*  Yf      = (float*)alloc((size_t)N * FIN_LD * sizeof(float));
  float*  score   = (float*)alloc((size_t)E * HH * sizeof(float));
  float*  mmax    = (float*)alloc((size_t)N * HH * sizeof(float));
  float*  denom   = (float*)alloc((size_t)N * HH * sizeof(float));
  float*  cp      = (float*)alloc((size_t)N * DD * sizeof(float));
  float*  aggf    = (float*)alloc((size_t)N * AG_LD * sizeof(float));
  float*  gsum    = (float*)alloc(2 * DD * sizeof(float));
  float*  gsumsq  = gsum + DD;
  (void)ws_size; (void)n_in; (void)out_size;

  const int BT = 256;
  bf16_t* wtcat[2] = { wtcat0, wtcat1 };
  float*  bcat[2]  = { bcat0, bcat1 };

  // ---- prep: bf16 activations, transposed bf16 weights, bias vectors ----
  f32_to_bf16<<<cdivl(N * DD, BT), BT, 0, stream>>>(x, xb, N * DD);
  for (int i = 0; i < 2; ++i) {
    const size_t wo = (size_t)i * KDIM * DD;
    prep_wt<<<cdivl((long)DD * KDIM, BT), BT, 0, stream>>>(Wl + wo,   wtcat[i],               KDIM, DD, DD);
    prep_wt<<<cdivl((long)DD * KDIM, BT), BT, 0, stream>>>(Wr + wo,   wtcat[i] + 256 * KDIM,  KDIM, DD, DD);
    prep_wt<<<cdivl((long)DD * KDIM, BT), BT, 0, stream>>>(Wlin + wo, wtcat[i] + 512 * KDIM,  KDIM, DD, DD);
  }
  prep_wt<<<cdivl((long)20 * KDIM, BT), BT, 0, stream>>>(fWl,   wtf,             KDIM, 20, 20);
  prep_wt<<<cdivl((long)20 * KDIM, BT), BT, 0, stream>>>(fWr,   wtf + 20 * KDIM, KDIM, 20, 20);
  prep_wt<<<cdivl((long)8  * KDIM, BT), BT, 0, stream>>>(fWlin, wtf + 40 * KDIM, KDIM, OUTC, 8);
  fill_f32<<<cdivl(2 * CAT_LD + FIN_LD, BT), BT, 0, stream>>>(bias, 2 * CAT_LD + FIN_LD, 0.0f);
  copy_f32<<<1, BT, 0, stream>>>(blin,       bcat0 + 512, DD);
  copy_f32<<<1, BT, 0, stream>>>(blin + DD,  bcat1 + 512, DD);
  copy_f32<<<1, BT, 0, stream>>>(fblin,      bcatf + 40,  OUTC);

  // ---- two GATv2 + linear-skip + GraphNorm + ELU layers ----
  for (int i = 0; i < 2; ++i) {
    const long groups = (N / 16) * ((CAT_LD / 16) / 4);
    gemm_bf16_wmma<4><<<cdivl(groups, 8), BT, 0, stream>>>(xb, wtcat[i], bcat[i], Ycat,
                                                           (int)N, KDIM, CAT_LD);
    fill_f32<<<cdivl(N * HH, BT), BT, 0, stream>>>(mmax,  N * HH, -3.0e38f);
    fill_f32<<<cdivl(N * HH, BT), BT, 0, stream>>>(denom, N * HH, 0.0f);
    fill_f32<<<cdivl(N * DD, BT), BT, 0, stream>>>(cp,    N * DD, 0.0f);
    fill_f32<<<1, BT, 0, stream>>>(gsum, 2 * DD, 0.0f);

    const long EH = E * HH;
    edge_score_w    <<<cdivl(E, 8), BT, 0, stream>>>(src, dst, Ycat, CAT_LD, 256,
                                                     att + (size_t)i * HH * CC, score, E);
    edge_segmax     <<<cdivl(EH, BT), BT, 0, stream>>>(dst, score, mmax, E, HH);
    edge_expsum     <<<cdivl(EH, BT), BT, 0, stream>>>(dst, score, mmax, denom, E, HH);
    edge_aggregate_w<<<cdivl(E, 8), BT, 0, stream>>>(src, dst, Ycat, CAT_LD,
                                                     score, denom, cp, E);

    gn_stats   <<<cdivl(N, GN_ROWS), BT, 0, stream>>>(cp, bconv + (size_t)i * DD, gsum, gsumsq, (int)N);
    gn_finalize<<<cdivl(N * DD, BT), BT, 0, stream>>>(cp, bconv + (size_t)i * DD, gsum, gsumsq,
                                                      gn_w + (size_t)i * DD, gn_b + (size_t)i * DD,
                                                      gn_ms + (size_t)i * DD,
                                                      Ycat + 512, CAT_LD, xb, (int)N);
  }

  // ---- final GATv2 (concat=False -> head mean) + linear + log_softmax ----
  {
    const long groups = (N / 16) * ((FIN_LD / 16) / 3);
    gemm_bf16_wmma<3><<<cdivl(groups, 8), BT, 0, stream>>>(xb, wtf, bcatf, Yf,
                                                           (int)N, KDIM, FIN_LD);
    fill_f32<<<cdivl(N * HH, BT), BT, 0, stream>>>(mmax,  N * HH, -3.0e38f);
    fill_f32<<<cdivl(N * HH, BT), BT, 0, stream>>>(denom, N * HH, 0.0f);
    fill_f32<<<cdivl(N * AG_LD, BT), BT, 0, stream>>>(aggf, N * AG_LD, 0.0f);

    const long EH = E * HH;
    edge_score    <<<cdivl(EH, BT), BT, 0, stream>>>(src, dst, Yf, FIN_LD, 0, 20,
                                                     fatt, score, E, HH, OUTC);
    edge_segmax   <<<cdivl(EH, BT), BT, 0, stream>>>(dst, score, mmax, E, HH);
    edge_expsum   <<<cdivl(EH, BT), BT, 0, stream>>>(dst, score, mmax, denom, E, HH);
    edge_aggregate<<<cdivl(EH, BT), BT, 0, stream>>>(src, dst, Yf, FIN_LD, 0,
                                                     score, denom, aggf, AG_LD, E, HH, OUTC);

    final_head<<<cdivl(N, BT), BT, 0, stream>>>(aggf, fbconv, Yf, out, (int)N);
  }
}
